// RelativeMultiHeadAttention_39118562132555
// MI455X (gfx1250) — matrix operand order repaired and checked by reading, not compiled, not run on hardware
//
#include <hip/hip_runtime.h>
#include <hip/hip_bf16.h>

// ---------------- CDNA5 WMMA types/helpers ----------------
typedef __attribute__((ext_vector_type(16))) _Float16 v16h;
typedef __attribute__((ext_vector_type(8)))  float    v8f;
typedef __attribute__((ext_vector_type(4)))  int      v4i;

union V16H { v16h h; uint4 u[2]; _Float16 e[16]; };

// load 16 f16 as two 16B chunks (A/B fragment halves per ISA layout)
__device__ __forceinline__ v16h ldh2(const _Float16* p0, const _Float16* p1) {
  V16H x;
  x.u[0] = *(const uint4*)p0;
  x.u[1] = *(const uint4*)p1;
  return x.h;
}
// convert 8+8 f32 -> packed v16h
__device__ __forceinline__ v16h cvt8x2(const float* p0, const float* p1) {
  V16H x;
#pragma unroll
  for (int i = 0; i < 8; ++i) x.e[i] = (_Float16)p0[i];
#pragma unroll
  for (int i = 0; i < 8; ++i) x.e[8 + i] = (_Float16)p1[i];
  return x.h;
}
__device__ __forceinline__ v16h cvt16(const float* p) { return cvt8x2(p, p + 8); }

__device__ __forceinline__ v8f wmma16(v16h a, v16h b, v8f c) {
  return __builtin_amdgcn_wmma_f32_16x16x32_f16(false, a, false, b, (short)0, c,
                                                false, false);
}

#define BB 2
#define NN 2048
#define CC 512
#define HH 16
#define DH 32
#define KTOP 16
#define SCALE 0.17677669529663687f /* 1/sqrt(32) */
#define MCHUNK 128                 /* K-columns staged per block in k_attn */

// CDNA5 async global->LDS path (ASYNCcnt). Guarded: falls back to direct
// global fragment loads if the builtin is not declared by this toolchain.
#if defined(__gfx1250__) && __has_builtin(__builtin_amdgcn_global_load_async_to_lds_b128)
#define ASYNC_OK 1
#else
#define ASYNC_OK 0
#endif

__device__ __forceinline__ void wait_async0() {
#if ASYNC_OK
#if __has_builtin(__builtin_amdgcn_s_wait_asynccnt)
  __builtin_amdgcn_s_wait_asynccnt(0);
#else
  asm volatile("s_wait_asynccnt 0" ::: "memory");
#endif
#endif
}

// stage MCHUNK*DH f16 (8 KB) from global to LDS; 128 threads x 4 x 16B lanes
__device__ __forceinline__ void stage_k(const _Float16* gsrc, _Float16* lbuf, int tid) {
#if ASYNC_OK
#pragma unroll
  for (int i = 0; i < (MCHUNK * DH * 2) / (128 * 16); ++i) {
    int off = tid * 16 + i * 128 * 16;
    __builtin_amdgcn_global_load_async_to_lds_b128(
        (__attribute__((address_space(1))) v4i*)((char*)gsrc + off),
        (__attribute__((address_space(3))) v4i*)((char*)lbuf + off), 0, 0);
  }
#else
  (void)gsrc; (void)lbuf; (void)tid;
#endif
}

// ---------------- K1: qkv = value @ w_qkv.T + b, pack q/k f16 [B,H,N,dh], v transposed ----------------
__global__ void k_qkv(const float* __restrict__ value, const float* __restrict__ wqkv,
                      const float* __restrict__ bqkv,
                      _Float16* __restrict__ qh, _Float16* __restrict__ kh,
                      _Float16* __restrict__ vT) {
  int wave = (blockIdx.x * blockDim.x + threadIdx.x) >> 5;
  int lane = threadIdx.x & 31;
  if (wave >= (BB * NN / 16) * (3 * CC / 16)) return;
  int jt = wave % (3 * CC / 16);
  int mt = wave / (3 * CC / 16);
  int row0 = mt * 16, j0 = jt * 16;
  int ln = lane & 15, lh = lane >> 4;
  int arow = row0 + ln;   // A row (global token)
  int jcol = j0 + ln;     // B col (output feature 0..1535)
  v8f acc = {};
#pragma unroll 4
  for (int kc = 0; kc < CC; kc += 32) {
    const float* ap = value + (size_t)arow * CC + kc + lh * 8;
    v16h a = cvt8x2(ap, ap + 16);
    v16h b = cvt8x2(wqkv + (size_t)jcol * CC + kc + lh * 8, wqkv + (size_t)jcol * CC + kc + lh * 8 + 16);
    acc = wmma16(a, b, acc);
  }
  float bias = bqkv[jcol];
  int which = jcol >> 9;          // 0=q 1=k 2=v
  int hh = (jcol >> 5) & 15;
  int d  = jcol & 31;
#pragma unroll
  for (int r = 0; r < 8; ++r) {
    int row = row0 + r + 8 * lh;
    int b_ = row >> 11, n_ = row & (NN - 1);
    _Float16 hv = (_Float16)(acc[r] + bias);
    if (which == 0)      qh[(((size_t)(b_ * HH + hh)) * NN + n_) * DH + d] = hv;
    else if (which == 1) kh[(((size_t)(b_ * HH + hh)) * NN + n_) * DH + d] = hv;
    else                 vT[(((size_t)(b_ * HH + hh)) * DH + d) * NN + n_] = hv;
  }
}

// ---------------- K2: z = (Q_full . K_full^T) * scale/H + gumbel(noise) ----------------
__global__ void k_logits(const _Float16* __restrict__ qh, const _Float16* __restrict__ kh,
                         const float* __restrict__ noise, float* __restrict__ z) {
  int wave = (blockIdx.x * blockDim.x + threadIdx.x) >> 5;
  int lane = threadIdx.x & 31;
  if (wave >= BB * (NN / 16) * (NN / 16)) return;
  int mt = wave & 127;
  int nt = (wave >> 7) & 127;
  int b  = wave >> 14;
  int n0 = nt * 16, m0 = mt * 16;
  int ln = lane & 15, lh = lane >> 4;
  v8f acc = {};
#pragma unroll 4
  for (int h = 0; h < HH; ++h) {  // K-chunk of 32 == one head's dh
    const _Float16* qp = qh + (((size_t)(b * HH + h)) * NN + n0 + ln) * DH + lh * 8;
    v16h a = ldh2(qp, qp + 16);
    const _Float16* kp = kh + (((size_t)(b * HH + h)) * NN + m0 + ln) * DH + lh * 8;
    v16h bb = ldh2(kp, kp + 16);
    acc = wmma16(a, bb, acc);
  }
  const float sc = SCALE / (float)HH;
#pragma unroll
  for (int r = 0; r < 8; ++r) {
    int row = n0 + r + 8 * lh;
    int col = m0 + ln;
    size_t idx = ((size_t)b * NN + row) * NN + col;
    float u = noise[idx];
    float g = -__logf(-__logf(u + 1e-9f) + 1e-9f);
    z[idx] = acc[r] * sc + g;
  }
}

// ---------------- K3: per-row 16th-largest of z (exact, bitwise binary search) ----------------
__global__ void k_topk(const float* __restrict__ z, float* __restrict__ thresh) {
  __shared__ unsigned ord[NN];
  __shared__ int scnt;
  int row = blockIdx.x;  // 0 .. B*N-1
  const float* zr = z + (size_t)row * NN;
  for (int i = threadIdx.x; i < NN; i += blockDim.x) {
    unsigned bts = __float_as_uint(zr[i]);
    ord[i] = bts ^ ((bts & 0x80000000u) ? 0xFFFFFFFFu : 0x80000000u);
  }
  __syncthreads();
  unsigned lo = 0u, hi = 0xFFFFFFFFu;
  for (int it = 0; it < 32; ++it) {
    unsigned span = hi - lo;
    unsigned mid = lo + (span >> 1) + (span & 1u);  // ceil midpoint, > lo when hi > lo
    if (threadIdx.x == 0) scnt = 0;
    __syncthreads();
    int local = 0;
    for (int i = threadIdx.x; i < NN; i += blockDim.x) local += (ord[i] >= mid) ? 1 : 0;
    atomicAdd(&scnt, local);
    __syncthreads();
    int c = scnt;
    __syncthreads();
    if (c >= KTOP) lo = mid; else hi = mid - 1;
  }
  if (threadIdx.x == 0) {
    unsigned o = lo;
    unsigned bts = (o & 0x80000000u) ? (o ^ 0x80000000u) : ~o;
    thresh[row] = __uint_as_float(bts);
  }
}

// ---------------- K4: per-head masked attention (2-pass flash, WMMA + async K staging) ----------------
// Block = 4 waves, all on the same (b,h): K tiles staged once per block into LDS.
__global__ void k_attn(const _Float16* __restrict__ qh, const _Float16* __restrict__ kh,
                       const _Float16* __restrict__ vT, const float* __restrict__ z,
                       const float* __restrict__ thresh, _Float16* __restrict__ attn_out) {
  __shared__ _Float16 kbuf[MCHUNK * DH];   // 8 KB shared K stage
  __shared__ _Float16 lds_p[4][16][32];    // 4 KB per-wave p-tile transpose staging
  int wid = threadIdx.x >> 5;
  int lane = threadIdx.x & 31;
  int tid = threadIdx.x;
  int wave = blockIdx.x * 4 + wid;
  int nt = wave & 127;
  int h  = (wave >> 7) & 15;
  int b  = wave >> 11;
  int n0 = nt * 16;
  int ln = lane & 15, lh = lane >> 4;
  v8f zacc = {};

  // q A-fragment (fixed for whole row-tile)
  const _Float16* qp = qh + (((size_t)(b * HH + h)) * NN + n0 + ln) * DH + lh * 8;
  v16h qa = ldh2(qp, qp + 16);
  const _Float16* kbase = kh + ((size_t)(b * HH + h)) * NN * DH;  // block-uniform
  const _Float16* vtb   = vT + ((size_t)(b * HH + h)) * DH * NN;

#if ASYNC_OK
#define KROW(mabs, mloc) (kbuf + (size_t)((mloc) + ln) * DH + lh * 8)
#else
#define KROW(mabs, mloc) (kbase + (size_t)((mabs) + ln) * DH + lh * 8)
#endif

  // ---- pass 1: row maxima of raw scores ----
  float rmax[8];
#pragma unroll
  for (int r = 0; r < 8; ++r) rmax[r] = -3.402823466e38f;
  for (int mc = 0; mc < NN; mc += MCHUNK) {
#if ASYNC_OK
    stage_k(kbase + (size_t)mc * DH, kbuf, tid);
    wait_async0();
    __syncthreads();
#endif
#pragma unroll
    for (int t = 0; t < MCHUNK / 16; ++t) {
      const _Float16* kp = KROW(mc + t * 16, t * 16);
      v8f s = wmma16(qa, ldh2(kp, kp + 16), zacc);
#pragma unroll
      for (int r = 0; r < 8; ++r) rmax[r] = fmaxf(rmax[r], s[r]);
    }
#if ASYNC_OK
    __syncthreads();  // protect kbuf before next chunk overwrites it
#endif
  }
#pragma unroll
  for (int m = 1; m <= 8; m <<= 1)
#pragma unroll
    for (int r = 0; r < 8; ++r) rmax[r] = fmaxf(rmax[r], __shfl_xor(rmax[r], m, 32));

  // ---- pass 2: exp, mask, accumulate p@v ----
  float th[8], rsum[8];
#pragma unroll
  for (int r = 0; r < 8; ++r) {
    th[r] = thresh[b * NN + n0 + r + 8 * lh];
    rsum[r] = 0.f;
  }
  v8f o0 = {}, o1 = {};
  for (int mc = 0; mc < NN; mc += MCHUNK) {
#if ASYNC_OK
    stage_k(kbase + (size_t)mc * DH, kbuf, tid);
    wait_async0();
    __syncthreads();
#endif
    // prefetch V stream for this chunk (global_prefetch_b8)
    __builtin_prefetch(vtb + (size_t)ln * NN + mc, 0, 1);
    __builtin_prefetch(vtb + (size_t)(16 + ln) * NN + mc, 0, 1);
#pragma unroll
    for (int sub = 0; sub < MCHUNK / 32; ++sub) {
      int mbase = mc + sub * 32;
#pragma unroll
      for (int half = 0; half < 2; ++half) {
        int m0 = mbase + half * 16;
        const _Float16* kp = KROW(m0, sub * 32 + half * 16);
        v8f s = wmma16(qa, ldh2(kp, kp + 16), zacc);
#pragma unroll
        for (int r = 0; r < 8; ++r) {
          int row = n0 + r + 8 * lh;
          int col = m0 + ln;
          float e = __expf(SCALE * (s[r] - rmax[r]));
          rsum[r] += e;  // softmax denominator over ALL columns
          float zc = z[((size_t)b * NN + row) * NN + col];
          float p = (zc >= th[r]) ? e : 0.f;
          lds_p[wid][r + 8 * lh][half * 16 + ln] = (_Float16)p;
        }
      }
      asm volatile("s_wait_dscnt 0" ::: "memory");  // wave-private LDS transpose
      __builtin_amdgcn_wave_barrier();
      // read p back in A-layout (16x32 f16)
      v16h pa = ldh2(&lds_p[wid][ln][lh * 8], &lds_p[wid][ln][16 + lh * 8]);
      // v B-fragments from transposed V: contiguous in m
      const _Float16* v0 = vtb + (size_t)ln * NN + mbase + lh * 8;
      const _Float16* v1 = vtb + (size_t)(16 + ln) * NN + mbase + lh * 8;
      o0 = wmma16(pa, ldh2(v0, v0 + 16), o0);
      o1 = wmma16(pa, ldh2(v1, v1 + 16), o1);
    }
#if ASYNC_OK
    __syncthreads();
#endif
  }
#pragma unroll
  for (int m = 1; m <= 8; m <<= 1)
#pragma unroll
    for (int r = 0; r < 8; ++r) rsum[r] += __shfl_xor(rsum[r], m, 32);
#pragma unroll
  for (int r = 0; r < 8; ++r) {
    int row = n0 + r + 8 * lh;
    float inv = 1.0f / rsum[r];
    size_t base = ((size_t)b * NN + row) * CC + h * DH;
    attn_out[base + ln]      = (_Float16)(o0[r] * inv);
    attn_out[base + 16 + ln] = (_Float16)(o1[r] * inv);
  }
#undef KROW
}

// ---------------- K5: out = (attn * mean(head_scores)) @ w_proj.T + b_proj ----------------
__global__ void k_proj(const _Float16* __restrict__ attn, const float* __restrict__ wproj,
                       const float* __restrict__ bproj, const float* __restrict__ hscores,
                       float* __restrict__ out) {
  int wave = (blockIdx.x * blockDim.x + threadIdx.x) >> 5;
  int lane = threadIdx.x & 31;
  if (wave >= (BB * NN / 16) * (CC / 16)) return;
  int jt = wave & 31;
  int mt = wave >> 5;
  int row0 = mt * 16, j0 = jt * 16;
  int ln = lane & 15, lh = lane >> 4;
  v8f acc = {};
#pragma unroll 4
  for (int kc = 0; kc < CC; kc += 32) {
    const _Float16* ap = attn + (size_t)(row0 + ln) * CC + kc + lh * 8;
    v16h a = ldh2(ap, ap + 16);
    v16h b = cvt8x2(wproj + (size_t)(j0 + ln) * CC + kc + lh * 8, wproj + (size_t)(j0 + ln) * CC + kc + lh * 8 + 16);
    acc = wmma16(a, b, acc);
  }
  float hs = 0.f;
#pragma unroll
  for (int i = 0; i < HH; ++i) hs += hscores[i];
  hs *= (1.0f / (float)HH);
  float bias = bproj[j0 + ln];
#pragma unroll
  for (int r = 0; r < 8; ++r) {
    int row = row0 + r + 8 * lh;
    out[(size_t)row * CC + j0 + ln] = acc[r] * hs + bias;
  }
}

// ---------------- launcher ----------------
extern "C" void kernel_launch(void* const* d_in, const int* in_sizes, int n_in,
                              void* d_out, int out_size, void* d_ws, size_t ws_size,
                              hipStream_t stream) {
  (void)in_sizes; (void)n_in; (void)out_size; (void)ws_size;
  const float* value = (const float*)d_in[2];
  const float* noise = (const float*)d_in[4];
  const float* wqkv  = (const float*)d_in[5];
  const float* bqkv  = (const float*)d_in[6];
  const float* wproj = (const float*)d_in[7];
  const float* bproj = (const float*)d_in[8];
  const float* hsc   = (const float*)d_in[9];

  char* w = (char*)d_ws;
  _Float16* qh = (_Float16*)(w + (size_t)0);            // 4 MB
  _Float16* kh = (_Float16*)(w + ((size_t)4  << 20));   // 4 MB
  _Float16* vT = (_Float16*)(w + ((size_t)8  << 20));   // 4 MB (transposed V)
  _Float16* at = (_Float16*)(w + ((size_t)12 << 20));   // 4 MB (attn output, f16)
  float*    z  = (float*)   (w + ((size_t)16 << 20));   // 32 MB (logits + gumbel)
  float*    th = (float*)   (w + ((size_t)48 << 20));   // 16 KB (per-row thresholds)
  float*    out = (float*)d_out;

  // K1: 256 M-tiles x 96 J-tiles = 24576 waves, 4 waves/block
  k_qkv   <<<6144, 128, 0, stream>>>(value, wqkv, bqkv, qh, kh, vT);
  // K2: 2 * 128 * 128 = 32768 waves
  k_logits<<<8192, 128, 0, stream>>>(qh, kh, noise, z);
  // K3: one block per row (4096 rows)
  k_topk  <<<4096, 256, 0, stream>>>(z, th);
  // K4: 2 * 16 * 128 = 4096 waves, 4 waves/block share (b,h)
  k_attn  <<<1024, 128, 0, stream>>>(qh, kh, vT, z, th, at);
  // K5: 256 * 32 = 8192 waves
  k_proj  <<<2048, 128, 0, stream>>>(at, wproj, bproj, hsc, out);
}
